// ProtFill_45423574122854
// MI455X (gfx1250) — matrix operand order repaired and checked by reading, not compiled, not run on hardware
//
#include <hip/hip_runtime.h>
#include <math.h>

#define BB 4
#define LL 2048
#define KK 30
#define NRBF 16
#define EDGE_F 128
#define EDGE_IN 272
#define KPAD 288
#define M_TOTAL (BB*LL*KK)   // 245760

typedef __attribute__((ext_vector_type(16))) __bf16 v16bf;
typedef __attribute__((ext_vector_type(8)))  __bf16 v8bf;
typedef __attribute__((ext_vector_type(8)))  float  v8f;

// atom codes: 0=N, 1=Ca, 2=Cb, 3=C   (15 pairs from the reference, i-atom / j-atom)
__constant__ int PA[15] = {0,1,2,3,3,3,0,0,2,0,1,2,1,2,1};
__constant__ int PB[15] = {0,1,2,0,1,2,1,2,1,3,3,3,0,0,2};

// ---------------- Kernel 1: pack atoms (N, Ca, Cb, C) per residue ----------------
__global__ void prep_atoms(const float* __restrict__ X, float* __restrict__ atoms) {
  int idx = blockIdx.x * blockDim.x + threadIdx.x;
  if (idx >= BB * LL) return;
  const float* r = X + (size_t)idx * 12;           // X[b,i,atom,xyz]; atoms: 0=N,1=C,2=Ca,3=O
  float Nx = r[0], Ny = r[1], Nz = r[2];
  float Cx = r[3], Cy = r[4], Cz = r[5];
  float Ax = r[6], Ay = r[7], Az = r[8];           // Ca
  float bx = Ax - Nx, by = Ay - Ny, bz = Az - Nz;  // b = Ca - N
  float cx = Cx - Ax, cy = Cy - Ay, cz = Cz - Az;  // c = C - Ca
  float ax = by * cz - bz * cy;
  float ay = bz * cx - bx * cz;
  float az = bx * cy - by * cx;
  float Bx = -0.58273431f * ax + 0.56802827f * bx - 0.54067466f * cx + Ax;
  float By = -0.58273431f * ay + 0.56802827f * by - 0.54067466f * cy + Ay;
  float Bz = -0.58273431f * az + 0.56802827f * bz - 0.54067466f * cz + Az;
  float* o = atoms + (size_t)idx * 16;
  o[0]  = Nx; o[1]  = Ny; o[2]  = Nz; o[3]  = 0.f;   // N
  o[4]  = Ax; o[5]  = Ay; o[6]  = Az; o[7]  = 0.f;   // Ca
  o[8]  = Bx; o[9]  = By; o[10] = Bz; o[11] = 0.f;   // Cb
  o[12] = Cx; o[13] = Cy; o[14] = Cz; o[15] = 0.f;   // C
}

// ---------------- Kernel 2: masked top-K (ascending, lower index wins ties) ------
__global__ __launch_bounds__(256) void topk_kernel(const float* __restrict__ X,
                                                   const float* __restrict__ mask,
                                                   int* __restrict__ Eidx,
                                                   float* __restrict__ Dn) {
  __shared__ float Drow[LL];
  __shared__ float rv[256];
  __shared__ int   ri[256];
  int row = blockIdx.x;              // b*L + i
  int b   = row / LL;
  int tid = threadIdx.x;
  const float* Ci = X + (size_t)row * 12 + 3;   // C atom = X[:,:,1,:]
  float cxx = Ci[0], cyy = Ci[1], czz = Ci[2];
  float mi = mask[row];
  float lmax = 0.f;
  for (int j = tid; j < LL; j += 256) {
    const float* Cj = X + (size_t)(b * LL + j) * 12 + 3;
    float dx = cxx - Cj[0], dy = cyy - Cj[1], dz = czz - Cj[2];
    float d  = sqrtf(dx * dx + dy * dy + dz * dz + 1e-6f);
    float m2 = mi * mask[b * LL + j];
    float Dv = m2 * d;
    Drow[j]  = Dv;
    lmax     = fmaxf(lmax, Dv);
  }
  rv[tid] = lmax;
  __syncthreads();
  for (int s = 128; s > 0; s >>= 1) {
    if (tid < s) rv[tid] = fmaxf(rv[tid], rv[tid + s]);
    __syncthreads();
  }
  float Dmax = rv[0];
  __syncthreads();
  for (int j = tid; j < LL; j += 256) {
    float m2 = mi * mask[b * LL + j];
    Drow[j] += (1.f - m2) * Dmax;    // D_adjust
  }
  __syncthreads();
  for (int k = 0; k < KK; ++k) {
    float bv = 3.4e38f; int bi = LL;
    for (int j = tid; j < LL; j += 256) {
      float v = Drow[j];
      if (v < bv || (v == bv && j < bi)) { bv = v; bi = j; }
    }
    rv[tid] = bv; ri[tid] = bi;
    __syncthreads();
    for (int s = 128; s > 0; s >>= 1) {
      if (tid < s) {
        float v2 = rv[tid + s]; int i2 = ri[tid + s];
        if (v2 < rv[tid] || (v2 == rv[tid] && i2 < ri[tid])) { rv[tid] = v2; ri[tid] = i2; }
      }
      __syncthreads();
    }
    if (tid == 0) {
      int sel = ri[0];
      Eidx[row * KK + k] = sel;
      Dn[row * KK + k]   = rv[0];
      Drow[sel] = 3.4e38f;
    }
    __syncthreads();
  }
}

// ---------------- Kernel 3: W_edge -> bf16, transposed [N=128][KPAD=288] ---------
__global__ void wtrans_kernel(const float* __restrict__ W_edge, __bf16* __restrict__ WT) {
  int t = blockIdx.x * 256 + threadIdx.x;
  if (t >= EDGE_F * KPAD) return;
  int n = t / KPAD, kp = t % KPAD;
  float v = (kp < EDGE_IN) ? W_edge[(size_t)kp * EDGE_F + n] : 0.f;
  WT[t] = (__bf16)v;
}

// ---------- Kernel 4: fused features -> LDS -> bf16 WMMA GEMM -> LayerNorm ------
__global__ __launch_bounds__(128) void edge_kernel(
    const float* __restrict__ atoms, const float* __restrict__ Dn,
    const int* __restrict__ Eidx, const int* __restrict__ ridx,
    const int* __restrict__ chains, const float* __restrict__ W_pos,
    const float* __restrict__ b_pos, const __bf16* __restrict__ WT,
    const float* __restrict__ gamma, const float* __restrict__ beta,
    float* __restrict__ Eout) {
  __shared__ __bf16 feat[4][16][KPAD];   // 36,864 bytes
  int wave  = threadIdx.x >> 5;
  int lane  = threadIdx.x & 31;
  int mtile = blockIdx.x * 4 + wave;     // 15360 tiles total
  int mbase = mtile * 16;

  // Phase 1: 16 edges x 18 groups-of-16 features = 288 tasks per wave (9 per lane)
  for (int s = 0; s < 9; ++s) {
    int t  = s * 32 + lane;
    int el = t / 18;
    int g  = t % 18;
    int edge = mbase + el;
    int bidx = edge / (LL * KK);
    int rem  = edge % (LL * KK);
    int i    = rem / KK;
    __bf16* out = &feat[wave][el][g * 16];
    if (g == 17) {                                   // K padding 272..287
      for (int f = 0; f < 16; ++f) out[f] = (__bf16)0.f;
    } else if (g == 0) {                             // positional embedding
      int j    = Eidx[edge];
      int off  = ridx[bidx * LL + i] - ridx[bidx * LL + j];
      int same = (chains[bidx * LL + i] == chains[bidx * LL + j]) ? 1 : 0;
      int d    = off + 32;
      if (d < 0)  d = 0;
      if (d > 64) d = 64;
      int dcls = same ? d : 65;
      const float* wp = W_pos + dcls * 16;
      for (int f = 0; f < 16; ++f) out[f] = (__bf16)(wp[f] + b_pos[f]);
    } else {                                         // one of 16 RBF sets
      float dval;
      if (g == 1) {
        dval = Dn[edge];                             // rbf(D_neighbors)
      } else {
        int p = g - 2;
        int j = Eidx[edge];
        const float* Ai = atoms + (size_t)(bidx * LL + i) * 16 + PA[p] * 4;
        const float* Bj = atoms + (size_t)(bidx * LL + j) * 16 + PB[p] * 4;
        float dx = Ai[0] - Bj[0], dy = Ai[1] - Bj[1], dz = Ai[2] - Bj[2];
        dval = sqrtf(dx * dx + dy * dy + dz * dz + 1e-6f);
      }
      for (int f = 0; f < 16; ++f) {
        float mu = 2.f + 1.33333333f * (float)f;     // linspace(2,22,16)
        float z  = (dval - mu) * 0.8f;               // sigma = 1.25
        out[f]   = (__bf16)__expf(-z * z);
      }
    }
  }
  __syncthreads();

  // Phase 2: 16x128 = A(16x288 bf16, LDS) x B(288x128 bf16, Wt in L2), f32 acc
  v8f acc[8];
  for (int n = 0; n < 8; ++n)
    for (int r = 0; r < 8; ++r) acc[n][r] = 0.f;
  const __bf16* arow = &feat[wave][lane & 15][0];
  int ahalf = (lane >> 4) * 8;     // A: lane<16 -> K{0..7,16..23}; lane>=16 -> K{8..15,24..31}
  int bhalf = (lane >> 4) * 8;     // B: lane<16 -> K{0..7,16..23}; lane>=16 -> K{8..15,24..31}
  for (int kt = 0; kt < 9; ++kt) {
    v8bf alo = *(const v8bf*)(arow + kt * 32 + ahalf);
    v8bf ahi = *(const v8bf*)(arow + kt * 32 + ahalf + 16);
    v16bf Af = __builtin_shufflevector(alo, ahi, 0,1,2,3,4,5,6,7,8,9,10,11,12,13,14,15);
    for (int n = 0; n < 8; ++n) {
      const __bf16* bcol = WT + (size_t)(n * 16 + (lane & 15)) * KPAD + kt * 32 + bhalf;
      v8bf blo = *(const v8bf*)(bcol);
      v8bf bhi = *(const v8bf*)(bcol + 16);
      v16bf Bf = __builtin_shufflevector(blo, bhi, 0,1,2,3,4,5,6,7,8,9,10,11,12,13,14,15);
      acc[n] = __builtin_amdgcn_wmma_f32_16x16x32_bf16(
          false, Af, false, Bf, (short)0, acc[n], false, false);
    }
  }

  // LayerNorm(128) per row via half-wave shuffles (rows 0-7 in lanes 0-15, 8-15 in 16-31)
  float p1[8], p2[8];
  for (int r = 0; r < 8; ++r) { p1[r] = 0.f; p2[r] = 0.f; }
  for (int n = 0; n < 8; ++n)
    for (int r = 0; r < 8; ++r) { float v = acc[n][r]; p1[r] += v; p2[r] += v * v; }
  for (int m = 1; m < 16; m <<= 1)
    for (int r = 0; r < 8; ++r) {
      p1[r] += __shfl_xor(p1[r], m, 32);
      p2[r] += __shfl_xor(p2[r], m, 32);
    }

  float gn[8], bn[8];
  for (int n = 0; n < 8; ++n) {
    int c = n * 16 + (lane & 15);
    gn[n] = gamma[c]; bn[n] = beta[c];
  }

  int radd = (lane >> 4) * 8;
  for (int r = 0; r < 8; ++r) {
    float mu  = p1[r] * (1.f / 128.f);
    float var = p2[r] * (1.f / 128.f) - mu * mu;
    float rs  = rsqrtf(var + 1e-5f);
    int rowg  = mbase + r + radd;
    float* orow = Eout + (size_t)rowg * EDGE_F;
    for (int n = 0; n < 8; ++n) {
      int c = n * 16 + (lane & 15);
      orow[c] = (acc[n][r] - mu) * rs * gn[n] + bn[n];
    }
  }
}

extern "C" void kernel_launch(void* const* d_in, const int* in_sizes, int n_in,
                              void* d_out, int out_size, void* d_ws, size_t ws_size,
                              hipStream_t stream) {
  const float* X      = (const float*)d_in[0];
  const float* mask   = (const float*)d_in[1];
  const int*   ridx   = (const int*)d_in[2];
  const int*   chains = (const int*)d_in[3];
  const float* W_pos  = (const float*)d_in[4];
  const float* b_pos  = (const float*)d_in[5];
  const float* W_edge = (const float*)d_in[6];
  const float* gamma  = (const float*)d_in[7];
  const float* beta   = (const float*)d_in[8];

  float* Eout   = (float*)d_out;
  int* EidxOut  = (int*)d_out + (size_t)M_TOTAL * EDGE_F;   // second tuple output

  float*  atoms = (float*)d_ws;                 // 131072 f32
  float*  Dn    = atoms + (size_t)BB * LL * 16; // 245760 f32
  __bf16* WT    = (__bf16*)(Dn + M_TOTAL);      // 36864 bf16

  prep_atoms<<<(BB * LL + 255) / 256, 256, 0, stream>>>(X, atoms);
  topk_kernel<<<BB * LL, 256, 0, stream>>>(X, mask, EidxOut, Dn);
  wtrans_kernel<<<(EDGE_F * KPAD + 255) / 256, 256, 0, stream>>>(W_edge, WT);
  edge_kernel<<<M_TOTAL / 16 / 4, 128, 0, stream>>>(atoms, Dn, EidxOut, ridx, chains,
                                                    W_pos, b_pos, WT, gamma, beta, Eout);
}
